// AdaptiveMixing_43928925503552
// MI455X (gfx1250) — compile-verified, matrix operand order checked by reading, not run on hardware
//
#include <hip/hip_runtime.h>
#include <hip/hip_bf16.h>
#include <cmath>
#include <cstdint>

// ---------------------------------------------------------------------------
// Problem constants (B=2, N=1024, P=32, C=256, OP=32, HID=128)
// ---------------------------------------------------------------------------
#define NTOK   2048            // B*N tokens
#define PP     32              // P
#define CDIM   256             // C
#define OPN    32              // OP
#define HIDN   128             // HID
#define CMSZ   (CDIM*CDIM)     // 65536 cm params per token
#define TOT    (CMSZ + OPN*PP) // 66560 total params per token
#define KD     (OPN*CDIM)      // 8192 contraction length of final projection

// LDS padding for bank-conflict-free access (64 banks x 4B):
#define CMR    40              // cm c-row pitch (halves): 80B -> lanes cover all banks
#define CMT    1288            // cm token pitch (halves): 2576B -> lane n starts bank 4n
#define XSR    40              // xs p-row pitch (halves): matches TDM pad (64B + 16B pad)
#define XST    (PP*XSR)        // xs token pitch = 1280 halves

typedef __attribute__((ext_vector_type(16))) __bf16 v16bf;
typedef __attribute__((ext_vector_type(8)))  float  v8f;
typedef __attribute__((ext_vector_type(4)))  unsigned int u32x4;
typedef __attribute__((ext_vector_type(8)))  int i32x8;
typedef __attribute__((ext_vector_type(4)))  int i32x4;

#if __has_builtin(__builtin_amdgcn_tensor_load_to_lds)
#define USE_TDM 1
#else
#define USE_TDM 0
#endif

union FragU { v16bf v; float4 q[2]; };
union PackU { __bf16 h[8]; float4 q; };
union F8U   { float4 q[2]; float f[8]; };

// WMMA bf16 16x16x32, f32 accumulate (CDNA5 wave32).
__device__ __forceinline__ v8f wmma_bf16(v16bf a, v16bf b, v8f c) {
  return __builtin_amdgcn_wmma_f32_16x16x32_bf16(false, a, false, b, (short)0, c,
                                                 false, false);
}

// Load one 16x32 bf16 A/B fragment. Layout per ISA 7.12.2 (16-bit A 16x32):
// lane l: r = l&15 (M for A / N for B), hi = l>>4;
// vector elems 0..7  = K in [8*hi    , 8*hi+7  ] -> one 16B load
// vector elems 8..15 = K in [16+8*hi , 16+8*hi+7] -> one 16B load
__device__ __forceinline__ v16bf load_frag(const __bf16* __restrict__ base, int stride) {
  const int l  = threadIdx.x & 31;
  const int r  = l & 15;
  const int hi = l >> 4;
  FragU u;
  u.q[0] = *reinterpret_cast<const float4*>(base + r * stride + 8 * hi);
  u.q[1] = *reinterpret_cast<const float4*>(base + r * stride + 16 + 8 * hi);
  return u.v;
}

__device__ __forceinline__ float gelu_exact(float x) {
  return 0.5f * x * (1.0f + erff(x * 0.70710678118654752f));
}

#if USE_TDM
// TDM: stage one xs chunk as a 3D tile (32 c x 32 p x 16 tokens, bf16) into LDS.
// Row pitch in LDS = 64B data + 16B pad = 80B (pad_interval=16 DWORDs, pad_amount=4),
// matching XSR=40 halves for conflict-free fragment reloads.
// This toolchain exposes the 6-arg builtin form:
//   (u32x4 g0, i32x8 g1, i32x4 g2, i32x4 g3, i32x8 g4, i32 cpol)
// Groups 0-3 carry the full D# for a <=5D non-gather tensor (ISA 8.3-8.6);
// the trailing i32x8 is zero-filled.
__device__ __forceinline__ void tdm_load_xs_chunk(const __bf16* gsrc, __bf16* ldst) {
  const unsigned long long ga = (unsigned long long)(uintptr_t)gsrc;
  u32x4 g0;
  g0[0] = 1u;                                   // count=1, user descriptor
  g0[1] = (unsigned)(uintptr_t)ldst;            // LDS byte offset (addr[31:0])
  g0[2] = (unsigned)ga;                         // global_addr[31:0]
  g0[3] = (unsigned)((ga >> 32) & 0x01FFFFFFu) | (2u << 30);  // addr[56:32]|type=2
  i32x8 g1;
  g1[0] = (1 << 16)        // data_size = 2B
        | (1 << 20)        // pad_enable
        | (3 << 22)        // pad_interval: every 16 DWORDs (one 64B row)
        | (3 << 25);       // pad_amount: 4 DWORDs (16B)
  g1[1] = 0;                                 // tensor_dim0[15:0]=0 (dim0=2^30)
  g1[2] = 0x4000;                            // tensor_dim0[31:16]; tensor_dim1 lo=0
  g1[3] = (32 << 16) | 0x4000;               // tile_dim0=32 | tensor_dim1[31:16]
  g1[4] = 32 | (16 << 16);                   // tile_dim1=32 (p) | tile_dim2=16 (tok)
  g1[5] = CDIM;                              // tensor_dim0_stride = 256 elems
  g1[6] = (int)(((unsigned)(PP * CDIM)) << 16); // tensor_dim1_stride lo16 = 8192
  g1[7] = 0;
  i32x4 g2; g2[0] = 0x40000000; g2[1] = 0; g2[2] = 0; g2[3] = 0; // dim2 big, 3D tile
  i32x4 g3; g3[0] = 0; g3[1] = 0; g3[2] = 0; g3[3] = 0;
  i32x8 g4; g4[0] = 0; g4[1] = 0; g4[2] = 0; g4[3] = 0;
  g4[4] = 0; g4[5] = 0; g4[6] = 0; g4[7] = 0;
  __builtin_amdgcn_tensor_load_to_lds(g0, g1, g2, g3, g4, 0);
}
#endif

// ---------------------------------------------------------------------------
// fp32 -> bf16 conversion (w2, x)
// ---------------------------------------------------------------------------
__global__ void convert_kernel(const float* __restrict__ src,
                               __bf16* __restrict__ dst, int n) {
  int i = blockIdx.x * blockDim.x + threadIdx.x;
  if (i < n) dst[i] = (__bf16)src[i];
}

// proj_w converted with K permutation: dst[e][d*32+o] = src[e][o*256+d]
__global__ void convert_proj_kernel(const float* __restrict__ src,
                                    __bf16* __restrict__ dst) {
  int i = blockIdx.x * blockDim.x + threadIdx.x;
  if (i < CDIM * KD) {
    int e = i >> 13, k2 = i & (KD - 1);
    int d = k2 >> 5, o = k2 & 31;
    dst[i] = (__bf16)src[(size_t)e * KD + o * CDIM + d];
  }
}

// b2 cm-part transposed to [d][c] (f32) so the GEMM1 epilogue bias load is
// two contiguous 16B loads per lane instead of 8 strided scalars.
__global__ void convert_b2t_kernel(const float* __restrict__ b2,
                                   float* __restrict__ B2T) {
  int i = blockIdx.x * blockDim.x + threadIdx.x;
  if (i < CMSZ) {
    int d = i >> 8, c = i & 255;
    B2T[i] = b2[(size_t)c * CDIM + d];
  }
}

// ---------------------------------------------------------------------------
// LayerNorm(query) then h = q @ w1^T + b1 ; store h as bf16 (trivial FLOPs)
// ---------------------------------------------------------------------------
__global__ void __launch_bounds__(256) prep_kernel(
    const float* __restrict__ query, const float* __restrict__ ln_w,
    const float* __restrict__ ln_b, const float* __restrict__ w1,
    const float* __restrict__ b1, __bf16* __restrict__ HB) {
  __shared__ float qn[CDIM];
  __shared__ float red[16];
  const int tok = blockIdx.x, t = threadIdx.x;
  float v = query[(size_t)tok * CDIM + t];
  float s = v, s2 = v * v;
#pragma unroll
  for (int o = 16; o > 0; o >>= 1) { s += __shfl_xor(s, o); s2 += __shfl_xor(s2, o); }
  if ((t & 31) == 0) { red[t >> 5] = s; red[8 + (t >> 5)] = s2; }
  __syncthreads();
  if (t == 0) {
    float a = 0.f, b = 0.f;
    for (int i = 0; i < 8; ++i) { a += red[i]; b += red[8 + i]; }
    red[0] = a * (1.0f / CDIM);
    red[8] = b * (1.0f / CDIM);
  }
  __syncthreads();
  const float mu = red[0];
  const float var = red[8] - mu * mu;
  qn[t] = (v - mu) * rsqrtf(var + 1e-6f) * ln_w[t] + ln_b[t];
  __syncthreads();
  if (t < HIDN) {
    float h = b1[t];
    const float* wr = w1 + (size_t)t * CDIM;
#pragma unroll 4
    for (int c = 0; c < CDIM; ++c) h += qn[c] * wr[c];
    HB[(size_t)tok * HIDN + t] = (__bf16)h;
  }
}

// ---------------------------------------------------------------------------
// SM[tok][o*32+p] = h_tok . w2[CMSZ + o*32+p] + b2  (WMMA: M=rows N=tokens K=128)
// ---------------------------------------------------------------------------
__global__ void __launch_bounds__(256) sm_kernel(
    const __bf16* __restrict__ W2B, const __bf16* __restrict__ HB,
    const float* __restrict__ b2, __bf16* __restrict__ SMB) {
  const int wave = threadIdx.x >> 5, lane = threadIdx.x & 31;
  const int nlo = lane & 15, mbase = (lane >> 4) * 8;
  const int job = blockIdx.x * 8 + wave;   // 8192 = 128 token-tiles x 64 row-tiles
  const int tok0 = (job >> 6) * 16;
  const int r0 = (job & 63) * 16;
  v8f acc = {0.f, 0.f, 0.f, 0.f, 0.f, 0.f, 0.f, 0.f};
#pragma unroll
  for (int k = 0; k < 4; ++k) {
    v16bf a = load_frag(W2B + (size_t)(CMSZ + r0) * HIDN + 32 * k, HIDN);
    v16bf b = load_frag(HB + (size_t)tok0 * HIDN + 32 * k, HIDN);
    acc = wmma_bf16(a, b, acc);
  }
  PackU p;
#pragma unroll
  for (int v = 0; v < 8; ++v)
    p.h[v] = (__bf16)(acc[v] + b2[CMSZ + r0 + mbase + v]);
  *reinterpret_cast<float4*>(SMB + (size_t)(tok0 + nlo) * 1024 + r0 + mbase) = p.q;
}

// ---------------------------------------------------------------------------
// Fused heavy kernel: per (16-token group, 32-wide d tile)
//   TDM double-buffers xs chunks into LDS (TENSORcnt path),
//   GEMM1 generates the cm chunk in LDS (never HBM), GEMM2 out1 += xs@cm,
//   epilogue gelu -> GEMM3 with sm -> gelu -> out2 bf16 [tok][d*32+o].
// ---------------------------------------------------------------------------
__global__ void __launch_bounds__(256) mix_kernel(
    const __bf16* __restrict__ XSB, const __bf16* __restrict__ W2B,
    const __bf16* __restrict__ HB, const float* __restrict__ B2T,
    const float* __restrict__ m_beta, const __bf16* __restrict__ SMB,
    const float* __restrict__ s_beta, __bf16* __restrict__ OUT2B) {
  __shared__ __align__(16) __bf16 lds_cm[16 * CMT];   // ~40.3 KB, cm chunk then g1
#if USE_TDM
  __shared__ __align__(16) __bf16 lds_xs[2][16 * XST]; // 2 x 40 KB double buffer
#endif

  const int g = blockIdx.x >> 3;       // token group (16 tokens)
  const int dt = blockIdx.x & 7;       // d tile (32 of 256 cols)
  const int wave = threadIdx.x >> 5;
  const int lane = threadIdx.x & 31;
  const int nlo = lane & 15;
  const int mbase = (lane >> 4) * 8;
  const int tok0 = g * 16;

#if USE_TDM
  if (wave == 0)
    tdm_load_xs_chunk(XSB + (size_t)tok0 * PP * CDIM, &lds_xs[0][0]);
#endif

  // H B-fragments (K = 0..127): resident in registers for the whole kernel.
  v16bf Bh[4];
#pragma unroll
  for (int k = 0; k < 4; ++k)
    Bh[k] = load_frag(HB + (size_t)tok0 * HIDN + 32 * k, HIDN);

  const v8f vzero = {0.f, 0.f, 0.f, 0.f, 0.f, 0.f, 0.f, 0.f};
  v8f acc1[2][2][2];  // [local token][p tile][d tile]
#pragma unroll
  for (int a = 0; a < 2; ++a)
#pragma unroll
    for (int b = 0; b < 2; ++b)
#pragma unroll
      for (int c = 0; c < 2; ++c) acc1[a][b][c] = vzero;

  for (int cc8 = 0; cc8 < 8; ++cc8) {  // 32-wide c chunks
#if USE_TDM
    if (wave == 0 && cc8 < 7)   // prefetch next xs chunk while GEMM1 runs
      tdm_load_xs_chunk(XSB + (size_t)tok0 * PP * CDIM + (cc8 + 1) * 32,
                        &lds_xs[(cc8 + 1) & 1][0]);
#endif
    // ---- GEMM1: cm[tok][d][c] chunk.  M = (d_local,c) rows of w2, N = tokens.
    for (int i = 0; i < 8; ++i) {
      const int mt = wave * 8 + i;
      const int d_local = mt >> 1;
      const int c_half = mt & 1;
      const int cbase = cc8 * 32 + c_half * 16;  // first c of this 16-row tile
      const int d = dt * 32 + d_local;
      const __bf16* abase = W2B + ((size_t)cbase * CDIM + d) * HIDN;
      v8f acc = vzero;
#pragma unroll
      for (int k = 0; k < 4; ++k) {
        v16bf a = load_frag(abase + 32 * k, CDIM * HIDN);  // row stride = one c step
        acc = wmma_bf16(a, Bh[k], acc);
      }
      if (cc8 < 7)  // warm next chunk's w2 rows (global_prefetch_b8)
        __builtin_prefetch((const void*)(abase + (size_t)32 * CDIM * HIDN +
                                         (size_t)nlo * CDIM * HIDN), 0, 1);
      F8U bv;  // transposed bias: two contiguous 16B loads
      const float* b2row = B2T + (size_t)d * CDIM + cbase + mbase;
      bv.q[0] = *reinterpret_cast<const float4*>(b2row);
      bv.q[1] = *reinterpret_cast<const float4*>(b2row + 4);
      PackU p;
#pragma unroll
      for (int v = 0; v < 8; ++v)
        p.h[v] = (__bf16)(acc[v] + bv.f[v]);
      // lds_cm [tok][d_local(pitch CMR)][c_local], conflict-free 16B stores
      *reinterpret_cast<float4*>(
          &lds_cm[nlo * CMT + d_local * CMR + c_half * 16 + mbase]) = p.q;
    }
#if USE_TDM
    if (wave == 0) {            // current xs chunk must have landed
      if (cc8 < 7) __builtin_amdgcn_s_wait_tensorcnt(1);
      else         __builtin_amdgcn_s_wait_tensorcnt(0);
    }
#endif
    __syncthreads();
    // ---- GEMM2: out1 += xs[:, c_chunk] @ cm_chunk, per token (2 per wave) ----
#pragma unroll
    for (int t2 = 0; t2 < 2; ++t2) {
      const int lt = wave * 2 + t2;
#if USE_TDM
      const __bf16* xrow = &lds_xs[cc8 & 1][lt * XST];
      v16bf a0 = load_frag(xrow, XSR);
      v16bf a1 = load_frag(xrow + 16 * XSR, XSR);
#else
      const size_t xrow = ((size_t)(tok0 + lt) * PP) * CDIM + cc8 * 32;
      v16bf a0 = load_frag(XSB + xrow, CDIM);
      v16bf a1 = load_frag(XSB + xrow + 16 * CDIM, CDIM);
#endif
#pragma unroll
      for (int dtl = 0; dtl < 2; ++dtl) {
        v16bf b = load_frag(&lds_cm[lt * CMT + dtl * 16 * CMR], CMR);
        acc1[t2][0][dtl] = wmma_bf16(a0, b, acc1[t2][0][dtl]);
        acc1[t2][1][dtl] = wmma_bf16(a1, b, acc1[t2][1][dtl]);
      }
    }
    __syncthreads();
  }

  // ---- epilogue: g1 = gelu(out1 + m_beta) -> lds_cm reused as [tok][d][p] ----
#pragma unroll
  for (int t2 = 0; t2 < 2; ++t2) {
    const int lt = wave * 2 + t2;
#pragma unroll
    for (int pt = 0; pt < 2; ++pt)
#pragma unroll
      for (int dtl = 0; dtl < 2; ++dtl) {
        const float mb = m_beta[dt * 32 + dtl * 16 + nlo];
        PackU p;
#pragma unroll
        for (int v = 0; v < 8; ++v)
          p.h[v] = (__bf16)gelu_exact(acc1[t2][pt][dtl][v] + mb);
        *reinterpret_cast<float4*>(
            &lds_cm[lt * CMT + (dtl * 16 + nlo) * CMR + pt * 16 + mbase]) = p.q;
      }
  }
  // Each wave reads back only its own tokens' region; in-wave DS ordering suffices.

  // ---- GEMM3: out2 = gelu(sm @ g1 + s_beta) -> OUT2B[tok][d*32+o] ----
#pragma unroll
  for (int t2 = 0; t2 < 2; ++t2) {
    const int lt = wave * 2 + t2;
    const int gt = tok0 + lt;
    v16bf sa0 = load_frag(SMB + (size_t)gt * 1024, 32);       // o tile 0
    v16bf sa1 = load_frag(SMB + (size_t)gt * 1024 + 512, 32); // o tile 1
#pragma unroll
    for (int dtl = 0; dtl < 2; ++dtl) {
      v16bf b = load_frag(&lds_cm[lt * CMT + dtl * 16 * CMR], CMR);
      v8f c0 = vzero, c1 = vzero;
      c0 = wmma_bf16(sa0, b, c0);
      c1 = wmma_bf16(sa1, b, c1);
      const int dcol = dt * 32 + dtl * 16 + nlo;
      __bf16* obase = OUT2B + (size_t)gt * KD + dcol * 32;
      PackU p;
#pragma unroll
      for (int v = 0; v < 8; ++v)
        p.h[v] = (__bf16)gelu_exact(c0[v] + s_beta[mbase + v]);
      *reinterpret_cast<float4*>(obase + mbase) = p.q;
#pragma unroll
      for (int v = 0; v < 8; ++v)
        p.h[v] = (__bf16)gelu_exact(c1[v] + s_beta[16 + mbase + v]);
      *reinterpret_cast<float4*>(obase + 16 + mbase) = p.q;
    }
  }
}

// ---------------------------------------------------------------------------
// Final projection: out = OUT2 (2048 x 8192) @ PROJB^T + proj_b  (fp32 out)
// ---------------------------------------------------------------------------
__global__ void __launch_bounds__(256) proj_kernel(
    const __bf16* __restrict__ OUT2B, const __bf16* __restrict__ PROJB,
    const float* __restrict__ proj_b, float* __restrict__ out) {
  const int wave = threadIdx.x >> 5, lane = threadIdx.x & 31;
  const int nlo = lane & 15, mbase = (lane >> 4) * 8;
  const int tok0 = blockIdx.x * 32 + (wave & 1) * 16;
  const int nb = wave >> 1;  // wave covers e tiles {nb, nb+4, nb+8, nb+12}
  const v8f vzero = {0.f, 0.f, 0.f, 0.f, 0.f, 0.f, 0.f, 0.f};
  v8f acc[4] = {vzero, vzero, vzero, vzero};
  for (int kc = 0; kc < KD / 32; ++kc) {
    v16bf a = load_frag(OUT2B + (size_t)tok0 * KD + kc * 32, KD);
#pragma unroll
    for (int j = 0; j < 4; ++j) {
      const int nt = nb + 4 * j;
      v16bf b = load_frag(PROJB + (size_t)(nt * 16) * KD + kc * 32, KD);
      acc[j] = wmma_bf16(a, b, acc[j]);
    }
  }
#pragma unroll
  for (int j = 0; j < 4; ++j) {
    const int e = (nb + 4 * j) * 16 + nlo;
    const float pb = proj_b[e];
#pragma unroll
    for (int v = 0; v < 8; ++v)
      out[(size_t)(tok0 + mbase + v) * CDIM + e] = acc[j][v] + pb;
  }
}

// ---------------------------------------------------------------------------
// Launch. Workspace (bf16 elems unless noted), ~93.4 MB total:
//   W2B 8,519,680 | PROJB 2,097,152 | XSB 16,777,216 | HB 262,144
//   SMB 2,097,152 | OUT2B 16,777,216 | B2T 65,536 f32
// ---------------------------------------------------------------------------
extern "C" void kernel_launch(void* const* d_in, const int* in_sizes, int n_in,
                              void* d_out, int out_size, void* d_ws, size_t ws_size,
                              hipStream_t stream) {
  const float* x      = (const float*)d_in[0];
  const float* query  = (const float*)d_in[1];
  const float* ln_w   = (const float*)d_in[2];
  const float* ln_b   = (const float*)d_in[3];
  const float* w1     = (const float*)d_in[4];
  const float* b1     = (const float*)d_in[5];
  const float* w2     = (const float*)d_in[6];
  const float* b2     = (const float*)d_in[7];
  const float* m_beta = (const float*)d_in[8];
  const float* s_beta = (const float*)d_in[9];
  const float* proj_w = (const float*)d_in[10];
  const float* proj_b = (const float*)d_in[11];

  __bf16* W2B   = (__bf16*)d_ws;
  __bf16* PROJB = W2B + (size_t)TOT * HIDN;
  __bf16* XSB   = PROJB + (size_t)CDIM * KD;
  __bf16* HB    = XSB + (size_t)NTOK * PP * CDIM;
  __bf16* SMB   = HB + (size_t)NTOK * HIDN;
  __bf16* OUT2B = SMB + (size_t)NTOK * OPN * PP;
  float*  B2T   = (float*)(OUT2B + (size_t)NTOK * KD);

  const int nW2 = TOT * HIDN;          // 8,519,680
  const int nXS = NTOK * PP * CDIM;    // 16,777,216
  convert_kernel<<<(nW2 + 255) / 256, 256, 0, stream>>>(w2, W2B, nW2);
  convert_proj_kernel<<<(CDIM * KD + 255) / 256, 256, 0, stream>>>(proj_w, PROJB);
  convert_kernel<<<(nXS + 255) / 256, 256, 0, stream>>>(x, XSB, nXS);
  convert_b2t_kernel<<<(CMSZ + 255) / 256, 256, 0, stream>>>(b2, B2T);
  prep_kernel<<<NTOK, 256, 0, stream>>>(query, ln_w, ln_b, w1, b1, HB);
  sm_kernel<<<1024, 256, 0, stream>>>(W2B, HB, b2, SMB);
  mix_kernel<<<1024, 256, 0, stream>>>(XSB, W2B, HB, B2T, m_beta, SMB, s_beta, OUT2B);
  proj_kernel<<<64, 256, 0, stream>>>(OUT2B, PROJB, proj_b, (float*)d_out);
}
